// MoELayer_26465588478459
// MI455X (gfx1250) — matrix operand order repaired and checked by reading, not compiled, not run on hardware
//
#include <hip/hip_runtime.h>

// ---------------------------------------------------------------- types
typedef __attribute__((ext_vector_type(16))) __bf16 v16bf;
typedef __attribute__((ext_vector_type(8)))  float  v8f;
typedef __attribute__((ext_vector_type(4)))  int    v4i;

union Frag16 { uint4 u[2]; v16bf v; };   // 32B = 16 bf16 = one WMMA A/B fragment per lane

// ---------------------------------------------------------------- problem sizes
constexpr int Bb = 4, Ss = 2048, Dd = 1024, Hh = 2048, Ee = 8;
constexpr int Tt = Bb * Ss;                  // 8192 tokens
constexpr int NP = Tt * 2;                   // 16384 (token,k) pairs
constexpr int M_TILE = 128;                  // rows per block tile
constexpr int N_TILE = 256;                  // cols per block tile
constexpr int MT_PER_E = Tt / M_TILE;        // 64 worst-case m-tiles per expert

// LDS staging (K_STAGE = 64, double buffered)
constexpr int A_LD = 64 + 8;                               // bf16 elems per A row (pad: bank-conflict-free)
constexpr int B_LD = 64 + 8;
constexpr int A_BYTES = M_TILE * A_LD * 2;                 // 18432
constexpr int B_BYTES = N_TILE * B_LD * 2;                 // 36864
constexpr int SMEM_PIDS = 2 * A_BYTES + 2 * B_BYTES;       // 110592
constexpr int SMEM_TOTAL = SMEM_PIDS + M_TILE * 4;         // 111104 B -> 2 blocks/WGP fit in 320KB

// ---------------------------------------------------------------- helpers
__device__ __forceinline__ unsigned short f2bf(float f) {
  unsigned int u = __float_as_uint(f);
  u += 0x7FFFu + ((u >> 16) & 1u);          // round-to-nearest-even
  return (unsigned short)(u >> 16);
}

// ------------------------------- gfx1250 async global->LDS copy (ASYNCcnt path)
#if defined(__gfx1250__) && __has_builtin(__builtin_amdgcn_global_load_async_to_lds_b128)
#define ASYNC_CP 1
#else
#define ASYNC_CP 0
#endif

__device__ __forceinline__ void cp_async_b128(const unsigned short* gsrc, unsigned short* ldst) {
#if ASYNC_CP
  __builtin_amdgcn_global_load_async_to_lds_b128(
      (__attribute__((address_space(1))) v4i*)gsrc,
      (__attribute__((address_space(3))) v4i*)ldst, 0, 0);
#else
  *(uint4*)ldst = *(const uint4*)gsrc;
#endif
}

__device__ __forceinline__ void cp_async_wait() {
#if ASYNC_CP
#if __has_builtin(__builtin_amdgcn_s_wait_asynccnt)
  __builtin_amdgcn_s_wait_asynccnt(0);
#else
  asm volatile("s_wait_asynccnt 0" ::: "memory");
#endif
#endif
}

// one K-stage: A half-row (64B) + B row (128B) per thread
__device__ __forceinline__ void stage_tile(const unsigned short* ag, unsigned short* al,
                                           const unsigned short* bg, unsigned short* bl) {
#if ASYNC_CP
#pragma unroll
  for (int c = 0; c < 4; ++c) cp_async_b128(ag + c * 8, al + c * 8);
#pragma unroll
  for (int c = 0; c < 8; ++c) cp_async_b128(bg + c * 8, bl + c * 8);
#else
  uint4 ta[4], tb[8];                        // two-phase: issue all loads, then all stores
#pragma unroll
  for (int c = 0; c < 4; ++c) ta[c] = ((const uint4*)ag)[c];
#pragma unroll
  for (int c = 0; c < 8; ++c) tb[c] = ((const uint4*)bg)[c];
#pragma unroll
  for (int c = 0; c < 4; ++c) ((uint4*)al)[c] = ta[c];
#pragma unroll
  for (int c = 0; c < 8; ++c) ((uint4*)bl)[c] = tb[c];
#endif
}

// ---------------------------------------------------------------- weight transpose + bf16 convert
// in: [E][R][C] fp32 -> out: [E][C][R] bf16
__global__ void __launch_bounds__(256) tconv_kernel(const float* __restrict__ in,
                                                    unsigned short* __restrict__ out,
                                                    int R, int C) {
  __shared__ float tile[32][33];
  const int tilesC = C >> 5, tilesR = R >> 5;
  const int per_e = tilesR * tilesC;
  const int e   = blockIdx.x / per_e;
  const int rem = blockIdx.x % per_e;
  const int r0 = (rem / tilesC) << 5;
  const int c0 = (rem % tilesC) << 5;
  const int j = threadIdx.x & 31;
  const int i = threadIdx.x >> 5;
  const float* ip = in + (size_t)e * R * C;
  unsigned short* op = out + (size_t)e * R * C;
#pragma unroll
  for (int ii = 0; ii < 4; ++ii) {
    int r = i + ii * 8;
    tile[r][j] = ip[(size_t)(r0 + r) * C + c0 + j];
  }
  __syncthreads();
#pragma unroll
  for (int ii = 0; ii < 4; ++ii) {
    int c = i + ii * 8;
    op[(size_t)(c0 + c) * R + r0 + j] = f2bf(tile[j][c]);
  }
}

// ---------------------------------------------------------------- x fp32 -> bf16
__global__ void __launch_bounds__(256) xconv_kernel(const float* __restrict__ x,
                                                    unsigned short* __restrict__ xbf) {
  int q = blockIdx.x * 256 + threadIdx.x;        // over T*D/4 float4s
  float4 v = ((const float4*)x)[q];
  uint2 p;
  p.x = (unsigned)f2bf(v.x) | ((unsigned)f2bf(v.y) << 16);
  p.y = (unsigned)f2bf(v.z) | ((unsigned)f2bf(v.w) << 16);
  ((uint2*)xbf)[q] = p;
}

// ---------------------------------------------------------------- router
__global__ void __launch_bounds__(256) router_kernel(const float* __restrict__ x,
                                                     const float* __restrict__ Wg,
                                                     float* __restrict__ gates,
                                                     int* __restrict__ counts,
                                                     int* __restrict__ lists,
                                                     float* __restrict__ gpair) {
  __shared__ float wgs[Ee * Dd];                 // Wg transposed [e][d], 32KB
  const int tid = threadIdx.x;
  for (int idx = tid; idx < Ee * Dd; idx += 256) {
    int d = idx >> 3, e = idx & 7;               // Wg is [D][E]
    wgs[e * Dd + d] = Wg[idx];
  }
  __syncthreads();

  const int lane = tid & 31;
  const int wid  = tid >> 5;
  const int t = blockIdx.x * 8 + wid;            // one wave per token
  float acc[Ee];
#pragma unroll
  for (int e = 0; e < Ee; ++e) acc[e] = 0.f;
  const float* xr = x + (size_t)t * Dd;
  for (int d = lane; d < Dd; d += 32) {
    float xd = xr[d];
#pragma unroll
    for (int e = 0; e < Ee; ++e) acc[e] += xd * wgs[e * Dd + d];
  }
#pragma unroll
  for (int e = 0; e < Ee; ++e)
    for (int off = 16; off > 0; off >>= 1)
      acc[e] += __shfl_xor(acc[e], off, 32);

  if (lane == 0) {
    int i0 = 0; float v0 = acc[0];
#pragma unroll
    for (int e = 1; e < Ee; ++e) if (acc[e] > v0) { v0 = acc[e]; i0 = e; }
    int i1 = (i0 == 0) ? 1 : 0; float v1 = acc[i1];
#pragma unroll
    for (int e = 0; e < Ee; ++e) if (e != i0 && acc[e] > v1) { v1 = acc[e]; i1 = e; }
    float e1 = __expf(v1 - v0);
    float g0 = 1.f / (1.f + e1);
    float g1 = e1 * g0;
#pragma unroll
    for (int e = 0; e < Ee; ++e)
      gates[(size_t)t * Ee + e] = (e == i0) ? g0 : ((e == i1) ? g1 : 0.f);
    gpair[t * 2 + 0] = g0;
    gpair[t * 2 + 1] = g1;
    int p0 = atomicAdd(&counts[i0], 1);
    lists[i0 * Tt + p0] = t * 2 + 0;             // pair id encodes (token,k)
    int p1 = atomicAdd(&counts[i1], 1);
    lists[i1 * Tt + p1] = t * 2 + 1;
  }
}

// ---------------------------------------------------------------- expert GEMM (one layer)
// C[128x256] tile per block; A rows gathered by pair-id; B = [E][N][K] bf16 weights.
// 8 waves = 2(M) x 4(N); each wave: 4x4 accumulators = 64x64 output.
template <bool OUT_EO>
__global__ void __launch_bounds__(256) expert_gemm_kernel(
    const unsigned short* __restrict__ Asrc, int shift, int lda,
    const unsigned short* __restrict__ Bw,   // [E][Ndim][Kdim]
    const float* __restrict__ bias,          // [E][Ndim]
    int Kdim, int Ndim, int ntiles,
    const int* __restrict__ lists, const int* __restrict__ counts,
    unsigned short* __restrict__ hOut,       // [NP][Ndim] bf16   (OUT_EO = false)
    float* __restrict__ eo) {                // [NP][Dd]   fp32   (OUT_EO = true)
  extern __shared__ char smem[];
  // runtime-computed LDS buffer pointers (no static-initializable aggregates!)
  auto abuf = [&](int i) -> unsigned short* {
    return (unsigned short*)(smem + i * A_BYTES);
  };
  auto bbuf = [&](int i) -> unsigned short* {
    return (unsigned short*)(smem + 2 * A_BYTES + i * B_BYTES);
  };
  int* pids = (int*)(smem + SMEM_PIDS);

  const int per_e = MT_PER_E * ntiles;
  const int e     = blockIdx.x / per_e;
  const int rem   = blockIdx.x % per_e;
  const int mtile = rem / ntiles;
  const int ntile = rem % ntiles;
  const int n_tok = counts[e];
  if (mtile * M_TILE >= n_tok) return;           // block-uniform early exit
  const int left = n_tok - mtile * M_TILE;
  const int rows = (left < M_TILE) ? left : M_TILE;
  const int tid = threadIdx.x;

  if (tid < M_TILE) {
    int slot = mtile * M_TILE + tid;
    pids[tid] = (tid < rows) ? lists[e * Tt + slot] : lists[e * Tt + mtile * M_TILE];
  }
  __syncthreads();

  // per-thread staging pointers (fixed across K-stages)
  const int ai = tid >> 1, ah = tid & 1;                       // 2 threads per A row
  const unsigned short* agp = Asrc + (size_t)((unsigned)pids[ai] >> shift) * lda + ah * 32;
  const int aoff = ai * A_LD + ah * 32;
  const unsigned short* bgp = Bw + ((size_t)e * Ndim + ntile * N_TILE + tid) * Kdim;
  const int boff = tid * B_LD;

  const int lane = tid & 31, wid = tid >> 5;
  const int wm = wid >> 2, wn = wid & 3;
  const int l15 = lane & 15, lhi = lane >> 4;

  v8f acc[4][4];
#pragma unroll
  for (int s = 0; s < 4; ++s)
#pragma unroll
    for (int q = 0; q < 4; ++q)
#pragma unroll
      for (int r = 0; r < 8; ++r) acc[s][q][r] = 0.f;

  // prologue: stage 0
  stage_tile(agp, abuf(0) + aoff, bgp, bbuf(0) + boff);
  cp_async_wait();
  __syncthreads();

  int cur = 0;
  for (int ks = 0; ks < Kdim; ks += 64) {
    if (ks + 64 < Kdim)                          // issue next stage, overlaps with WMMAs below
      stage_tile(agp + ks + 64, abuf(cur ^ 1) + aoff,
                 bgp + ks + 64, bbuf(cur ^ 1) + boff);
    const unsigned short* ab = abuf(cur);
    const unsigned short* bb = bbuf(cur);
#pragma unroll
    for (int k2 = 0; k2 < 64; k2 += 32) {
      Frag16 af[4];
#pragma unroll
      for (int s = 0; s < 4; ++s) {              // A 16x32 layout: K = base..base+7, base+16..base+23
        const unsigned short* ap = ab + (wm * 64 + s * 16 + l15) * A_LD + k2 + lhi * 8;
        af[s].u[0] = *(const uint4*)ap;
        af[s].u[1] = *(const uint4*)(ap + 16);
      }
#pragma unroll
      for (int q = 0; q < 4; ++q) {              // B: lane -> column, K = base..base+7, base+16..base+23 (same as A)
        Frag16 bfr;
        const unsigned short* bp = bb + (wn * 64 + q * 16 + l15) * B_LD + k2 + lhi * 8;
        bfr.u[0] = *(const uint4*)bp;
        bfr.u[1] = *(const uint4*)(bp + 16);
#pragma unroll
        for (int s = 0; s < 4; ++s)
          acc[s][q] = __builtin_amdgcn_wmma_f32_16x16x32_bf16(
              false, af[s].v, false, bfr.v, (short)0, acc[s][q], false, false);
      }
    }
    cp_async_wait();                             // next-stage copies done (usually already)
    __syncthreads();                             // all waves finished reading `cur`
    cur ^= 1;
  }

  // epilogue
#pragma unroll
  for (int q = 0; q < 4; ++q) {
    const int col = ntile * N_TILE + wn * 64 + q * 16 + l15;
    const float bv = bias[(size_t)e * Ndim + col];
#pragma unroll
    for (int s = 0; s < 4; ++s) {
      const int rbase = wm * 64 + s * 16 + lhi * 8;
#pragma unroll
      for (int r = 0; r < 8; ++r) {
        const int row = rbase + r;
        if (row < rows) {
          float v = acc[s][q][r] + bv;
          if (OUT_EO) {
            eo[(size_t)pids[row] * Dd + col] = v;
          } else {
            v = v > 0.f ? v : 0.f;               // relu
            hOut[(size_t)pids[row] * Ndim + col] = f2bf(v);
          }
        }
      }
    }
  }
}

// ---------------------------------------------------------------- combine: out = g0*eo0 + g1*eo1
__global__ void __launch_bounds__(256) combine_kernel(const float* __restrict__ eo,
                                                      const float* __restrict__ gpair,
                                                      float* __restrict__ out) {
  int q = blockIdx.x * 256 + threadIdx.x;        // Tt * (Dd/4)
  int t = q >> 8, c4 = q & 255;
  float g0 = gpair[t * 2 + 0];
  float g1 = gpair[t * 2 + 1];
  float4 a = ((const float4*)eo)[(size_t)(t * 2 + 0) * 256 + c4];
  float4 b = ((const float4*)eo)[(size_t)(t * 2 + 1) * 256 + c4];
  float4 o;
  o.x = g0 * a.x + g1 * b.x;
  o.y = g0 * a.y + g1 * b.y;
  o.z = g0 * a.z + g1 * b.z;
  o.w = g0 * a.w + g1 * b.w;
  ((float4*)out)[q] = o;
}

// ---------------------------------------------------------------- load-balance loss (deterministic)
__global__ void __launch_bounds__(256) loss_kernel(const float* __restrict__ gates,
                                                   const int* __restrict__ counts,
                                                   float* __restrict__ lb) {
  __shared__ float red[Ee][256];
  const int tid = threadIdx.x;
  float s[Ee];
#pragma unroll
  for (int e = 0; e < Ee; ++e) s[e] = 0.f;
  for (int t = tid; t < Tt; t += 256)
#pragma unroll
    for (int e = 0; e < Ee; ++e) s[e] += gates[(size_t)t * Ee + e];
#pragma unroll
  for (int e = 0; e < Ee; ++e) red[e][tid] = s[e];
  __syncthreads();
  if (tid < Ee) {
    float tot = 0.f;
    for (int i = 0; i < 256; ++i) tot += red[tid][i];
    red[tid][0] = tot;
  }
  __syncthreads();
  if (tid == 0) {
    float v = 0.f;
    for (int e = 0; e < Ee; ++e)
      v += (red[e][0] / (float)Tt) * ((float)counts[e] / (float)Tt);
    *lb = v * (float)Ee;
  }
}

// ---------------------------------------------------------------- host
extern "C" void kernel_launch(void* const* d_in, const int* in_sizes, int n_in,
                              void* d_out, int out_size, void* d_ws, size_t ws_size,
                              hipStream_t stream) {
  const float* x  = (const float*)d_in[0];
  const float* Wg = (const float*)d_in[1];
  const float* W1 = (const float*)d_in[2];
  const float* b1 = (const float*)d_in[3];
  const float* W2 = (const float*)d_in[4];
  const float* b2 = (const float*)d_in[5];
  const float* W3 = (const float*)d_in[6];
  const float* b3 = (const float*)d_in[7];

  float* out   = (float*)d_out;                   // [T][D]
  float* gates = out + (size_t)Tt * Dd;           // [T][E]
  float* lb    = gates + (size_t)Tt * Ee;         // scalar

  char* ws = (char*)d_ws;
  size_t off = 0;
  auto alloc = [&](size_t bytes) {
    void* p = ws + off;
    off = (off + bytes + 255) & ~(size_t)255;
    return p;
  };
  unsigned short* W1t = (unsigned short*)alloc((size_t)Ee * Hh * Dd * 2);  // [E][H][D] bf16
  unsigned short* W2t = (unsigned short*)alloc((size_t)Ee * Hh * Hh * 2);  // [E][H][H] bf16
  unsigned short* W3t = (unsigned short*)alloc((size_t)Ee * Dd * Hh * 2);  // [E][D][H] bf16
  unsigned short* xbf = (unsigned short*)alloc((size_t)Tt * Dd * 2);       // [T][D]  bf16
  unsigned short* h1  = (unsigned short*)alloc((size_t)NP * Hh * 2);       // [NP][H] bf16
  unsigned short* h2  = (unsigned short*)alloc((size_t)NP * Hh * 2);       // [NP][H] bf16
  float* eo    = (float*)alloc((size_t)NP * Dd * 4);                       // [NP][D] fp32
  int*   lists = (int*)  alloc((size_t)Ee * Tt * 4);
  int*   counts= (int*)  alloc(256);
  float* gpair = (float*)alloc((size_t)Tt * 2 * 4);

  (void)hipMemsetAsync(counts, 0, Ee * sizeof(int), stream);
  (void)hipFuncSetAttribute((const void*)expert_gemm_kernel<false>,
                            hipFuncAttributeMaxDynamicSharedMemorySize, SMEM_TOTAL);
  (void)hipFuncSetAttribute((const void*)expert_gemm_kernel<true>,
                            hipFuncAttributeMaxDynamicSharedMemorySize, SMEM_TOTAL);

  tconv_kernel<<<Ee * (Dd / 32) * (Hh / 32), 256, 0, stream>>>(W1, W1t, Dd, Hh);
  tconv_kernel<<<Ee * (Hh / 32) * (Hh / 32), 256, 0, stream>>>(W2, W2t, Hh, Hh);
  tconv_kernel<<<Ee * (Hh / 32) * (Dd / 32), 256, 0, stream>>>(W3, W3t, Hh, Dd);
  xconv_kernel<<<Tt * Dd / 4 / 256, 256, 0, stream>>>(x, xbf);

  router_kernel<<<Tt / 8, 256, 0, stream>>>(x, Wg, gates, counts, lists, gpair);

  // layer1: h1 = relu(x @ W1 + b1)   A = xbf rows (pair>>1), K=1024, N=2048
  expert_gemm_kernel<false><<<Ee * MT_PER_E * 8, 256, SMEM_TOTAL, stream>>>(
      xbf, 1, Dd, W1t, b1, Dd, Hh, 8, lists, counts, h1, nullptr);
  // layer2: h2 = relu(h1 @ W2 + b2)  A = h1 rows (pair), K=2048, N=2048
  expert_gemm_kernel<false><<<Ee * MT_PER_E * 8, 256, SMEM_TOTAL, stream>>>(
      h1, 0, Hh, W2t, b2, Hh, Hh, 8, lists, counts, h2, nullptr);
  // layer3: eo = h2 @ W3 + b3        A = h2 rows (pair), K=2048, N=1024
  expert_gemm_kernel<true><<<Ee * MT_PER_E * 4, 256, SMEM_TOTAL, stream>>>(
      h2, 0, Hh, W3t, b3, Hh, Dd, 4, lists, counts, nullptr, eo);

  combine_kernel<<<Tt, 256, 0, stream>>>(eo, gpair, out);
  loss_kernel<<<1, 256, 0, stream>>>(gates, counts, lb);

  (void)in_sizes; (void)n_in; (void)out_size; (void)ws_size;
}
